// ParallelFcCoembeddingSigmoidWithAttention_1872605741177
// MI455X (gfx1250) — matrix operand order repaired and checked by reading, not compiled, not run on hardware
//
#include <hip/hip_runtime.h>
#include <cstdint>
#include <cstddef>

// ---------------------------------------------------------------------------
// Types
// ---------------------------------------------------------------------------
typedef __bf16 bf16_t;
typedef __attribute__((ext_vector_type(16))) __bf16 v16bf;
typedef __attribute__((ext_vector_type(8)))  float  v8f;
typedef __attribute__((ext_vector_type(4)))  float  v4f;

// Problem constants (match reference)
#define D_LAT   1024
#define NHEAD   8
#define DHEAD   128
#define BATCH   128

// GEMM tiling
#define BM 128
#define BN 128
#define BK 32
#define LDSK 48   // padded K stride in LDS (96B rows, 16B aligned, conflict-skewed)

// ---------------------------------------------------------------------------
// fp32 -> bf16 bulk convert (weights)
// ---------------------------------------------------------------------------
__global__ __launch_bounds__(256)
void cvt_f32_bf16(const float* __restrict__ src, bf16_t* __restrict__ dst, int n) {
  int i = blockIdx.x * blockDim.x + threadIdx.x;
  int stride = gridDim.x * blockDim.x;
  for (; i < n; i += stride) dst[i] = (bf16_t)src[i];
}

// ---------------------------------------------------------------------------
// h[b*F+f, d] = relu(x[b,f] * W[f,d] + bvec[f,d])  -> bf16
// ---------------------------------------------------------------------------
__global__ __launch_bounds__(256)
void feature_embed_relu(const float* __restrict__ x, int xStride, int F,
                        const float* __restrict__ W, const float* __restrict__ bvec,
                        bf16_t* __restrict__ h) {
  int row = blockIdx.x;            // b*F + f
  int b = row / F, f = row - b * F;
  float xv = x[(size_t)b * xStride + f];
  const float* Wr = W    + (size_t)f * D_LAT;
  const float* br = bvec + (size_t)f * D_LAT;
  bf16_t* hr = h + (size_t)row * D_LAT;
  for (int d = threadIdx.x; d < D_LAT; d += blockDim.x) {
    float v = fmaf(xv, Wr[d], br[d]);
    hr[d] = (bf16_t)fmaxf(v, 0.0f);
  }
}

// ---------------------------------------------------------------------------
// Strided-batched WMMA GEMM:  C = alpha * (A x B) + bias
//   A:  M x K row-major bf16, leading dim lda
//   B:  if BTRANS: stored as N x K rows (ldb over K)  -> C = A * B^T(stored)
//       else:      stored as K x N rows (ldb over N)  -> C = A * B
//   batch z -> (bi = z / nhDiv, hi = z % nhDiv); per-operand strides in elems.
// Block tile 128x128, 8 waves (4 along M x 2 along N), wave tile 32x64,
// inner op v_wmma_f32_16x16x32_bf16. All dims must divide tiles (they do here).
// ---------------------------------------------------------------------------
template <typename OutT, bool BTRANS>
__global__ __launch_bounds__(256)
void gemm_wmma(const bf16_t* __restrict__ A, int lda, long aSB, long aSH,
               const bf16_t* __restrict__ Bm, int ldb, long bSB, long bSH,
               OutT* __restrict__ C, int ldc, long cSB, long cSH,
               const float* __restrict__ bias, float alpha,
               int M, int N, int K, int nhDiv) {
  __shared__ bf16_t lds_a[BM][LDSK];
  __shared__ bf16_t lds_b[BN][LDSK];

  const int z  = blockIdx.z;
  const int bi = z / nhDiv;
  const int hi = z - bi * nhDiv;
  A  += (long)bi * aSB + (long)hi * aSH;
  Bm += (long)bi * bSB + (long)hi * bSH;
  C  += (long)bi * cSB + (long)hi * cSH;

  const int mBlock = blockIdx.y * BM;
  const int nBlock = blockIdx.x * BN;

  const int tid   = threadIdx.x;
  const int wave  = tid >> 5;
  const int lane  = tid & 31;
  const int half  = lane >> 4;     // 0: lanes 0-15, 1: lanes 16-31
  const int l16   = lane & 15;
  const int waveM = (wave & 3) * 32;   // 4 waves along M
  const int waveN = (wave >> 2) * 64;  // 2 waves along N

  v8f acc[2][4];
#pragma unroll
  for (int mt = 0; mt < 2; ++mt)
#pragma unroll
    for (int nt = 0; nt < 4; ++nt) acc[mt][nt] = (v8f)0.0f;

  // cooperative loader indices
  const int arow = tid & 127;      // 128 rows
  const int ahs  = tid >> 7;       // which 16-half chunk of the 32-wide K slab

  for (int k0 = 0; k0 < K; k0 += BK) {
    // ---- stage A tile (BM x BK) : always row-major M x K ----
    {
      const bf16_t* gp = A + (size_t)(mBlock + arow) * lda + k0 + ahs * 16;
      v4f* dp = (v4f*)&lds_a[arow][ahs * 16];
      const v4f* sp = (const v4f*)gp;
      dp[0] = sp[0];
      dp[1] = sp[1];
      if (k0 + BK < K) __builtin_prefetch(gp + BK, 0, 0);  // global_prefetch next slab
    }
    // ---- stage B tile (stored as [n][k] in LDS) ----
    if (BTRANS) {
      const bf16_t* gp = Bm + (size_t)(nBlock + arow) * ldb + k0 + ahs * 16;
      v4f* dp = (v4f*)&lds_b[arow][ahs * 16];
      const v4f* sp = (const v4f*)gp;
      dp[0] = sp[0];
      dp[1] = sp[1];
      if (k0 + BK < K) __builtin_prefetch(gp + BK, 0, 0);
    } else {
      // B stored K x N: transpose 32x128 slab into lds_b[n][k]
      const int kk   = tid & 31;
      const int nSub = tid >> 5;   // 0..7, 16 columns each
      union { v4f f4[2]; bf16_t e[16]; } tmp;
      const v4f* sp = (const v4f*)(Bm + (size_t)(k0 + kk) * ldb + nBlock + nSub * 16);
      tmp.f4[0] = sp[0];
      tmp.f4[1] = sp[1];
#pragma unroll
      for (int j = 0; j < 16; ++j) lds_b[nSub * 16 + j][kk] = tmp.e[j];
    }
    __syncthreads();

    // ---- fragments per ISA 7.12.2 16-bit layouts ----
    union FragU { v16bf v; v4f f4[2]; };
    FragU aF[2], bF[4];
#pragma unroll
    for (int mt = 0; mt < 2; ++mt) {
      const int m = waveM + mt * 16 + l16;
      // lanes 0-15: K {0..7, 16..23}; lanes 16-31: K {8..15, 24..31}
      aF[mt].f4[0] = *(const v4f*)&lds_a[m][half * 8];
      aF[mt].f4[1] = *(const v4f*)&lds_a[m][16 + half * 8];
    }
#pragma unroll
    for (int nt = 0; nt < 4; ++nt) {
      const int n = waveN + nt * 16 + l16;
      // lanes 0-15: K {0..7, 16..23}; lanes 16-31: K {8..15, 24..31}
      bF[nt].f4[0] = *(const v4f*)&lds_b[n][half * 8];
      bF[nt].f4[1] = *(const v4f*)&lds_b[n][16 + half * 8];
    }

#pragma unroll
    for (int mt = 0; mt < 2; ++mt)
#pragma unroll
      for (int nt = 0; nt < 4; ++nt)
        acc[mt][nt] = __builtin_amdgcn_wmma_f32_16x16x32_bf16(
            false, aF[mt].v, false, bF[nt].v, (short)0, acc[mt][nt], false, false);

    __syncthreads();
  }

  // ---- epilogue: C/D layout -> element (m = r + 8*half, n = l16) per tile ----
#pragma unroll
  for (int mt = 0; mt < 2; ++mt) {
#pragma unroll
    for (int nt = 0; nt < 4; ++nt) {
      const int col = nBlock + waveN + nt * 16 + l16;
      const float bv = bias ? bias[col] : 0.0f;
#pragma unroll
      for (int r = 0; r < 8; ++r) {
        const int row = mBlock + waveM + mt * 16 + half * 8 + r;
        const float v = alpha * acc[mt][nt][r] + bv;
        C[(size_t)row * ldc + col] = (OutT)v;
      }
    }
  }
}

// ---------------------------------------------------------------------------
// Row softmax: f32 scores (rows of length F<=256) -> bf16 attn
// one block per row
// ---------------------------------------------------------------------------
__global__ __launch_bounds__(256)
void softmax_rows(const float* __restrict__ scores, bf16_t* __restrict__ attn, int F) {
  __shared__ float red[256];
  const long row = blockIdx.x;
  const float* s = scores + row * F;
  bf16_t* a = attn + row * F;
  const int t = threadIdx.x;
  float v = (t < F) ? s[t] : -3.4e38f;
  red[t] = v; __syncthreads();
  for (int off = 128; off > 0; off >>= 1) {
    if (t < off) red[t] = fmaxf(red[t], red[t + off]);
    __syncthreads();
  }
  const float mx = red[0]; __syncthreads();
  const float e = (t < F) ? __expf(v - mx) : 0.0f;
  red[t] = e; __syncthreads();
  for (int off = 128; off > 0; off >>= 1) {
    if (t < off) red[t] += red[t + off];
    __syncthreads();
  }
  const float inv = 1.0f / red[0];
  if (t < F) a[t] = (bf16_t)(e * inv);
}

// ---------------------------------------------------------------------------
// LayerNorm over D=1024 per row + accumulate into per-batch sum via atomics
// one block per (b,f) row; acc must be pre-zeroed
// ---------------------------------------------------------------------------
__global__ __launch_bounds__(256)
void layernorm_sum(const float* __restrict__ o, const float* __restrict__ g,
                   const float* __restrict__ beta, float* __restrict__ acc, int F) {
  __shared__ float red[256];
  const long row = blockIdx.x;
  const int b = (int)(row / F);
  const float* orow = o + row * D_LAT;
  const int t = threadIdx.x;
  float x[4]; float s = 0.0f;
#pragma unroll
  for (int j = 0; j < 4; ++j) { x[j] = orow[t * 4 + j]; s += x[j]; }
  red[t] = s; __syncthreads();
  for (int off = 128; off > 0; off >>= 1) { if (t < off) red[t] += red[t + off]; __syncthreads(); }
  const float mu = red[0] * (1.0f / D_LAT); __syncthreads();
  float s2 = 0.0f;
#pragma unroll
  for (int j = 0; j < 4; ++j) { const float d = x[j] - mu; s2 += d * d; }
  red[t] = s2; __syncthreads();
  for (int off = 128; off > 0; off >>= 1) { if (t < off) red[t] += red[t + off]; __syncthreads(); }
  const float rstd = rsqrtf(red[0] * (1.0f / D_LAT) + 1e-5f);
#pragma unroll
  for (int j = 0; j < 4; ++j) {
    const int d = t * 4 + j;
    const float val = (x[j] - mu) * rstd * g[d] + beta[d];
    atomicAdd(&acc[(size_t)b * D_LAT + d], val);
  }
}

// ---------------------------------------------------------------------------
// out[b] = sigmoid( dot(dp,tp) / max(||dp||*||tp||, 1e-8) )
// ---------------------------------------------------------------------------
__global__ __launch_bounds__(256)
void cosine_sigmoid(const float* __restrict__ dp, const float* __restrict__ tp,
                    float* __restrict__ out) {
  __shared__ float rd[256], ra[256], rb[256];
  const int b = blockIdx.x, t = threadIdx.x;
  const float* dpr = dp + (size_t)b * D_LAT;
  const float* tpr = tp + (size_t)b * D_LAT;
  float sd = 0.0f, sa = 0.0f, sb = 0.0f;
#pragma unroll
  for (int j = 0; j < 4; ++j) {
    const float a = dpr[t * 4 + j], c = tpr[t * 4 + j];
    sd += a * c; sa += a * a; sb += c * c;
  }
  rd[t] = sd; ra[t] = sa; rb[t] = sb; __syncthreads();
  for (int off = 128; off > 0; off >>= 1) {
    if (t < off) { rd[t] += rd[t + off]; ra[t] += ra[t + off]; rb[t] += rb[t + off]; }
    __syncthreads();
  }
  if (t == 0) {
    const float denom = fmaxf(sqrtf(ra[0] * rb[0]), 1e-8f);
    const float c = rd[0] / denom;
    out[b] = 1.0f / (1.0f + __expf(-c));
  }
}

// ---------------------------------------------------------------------------
// Workspace layout (bytes, sized for the larger drug path, reused for target)
// ---------------------------------------------------------------------------
static constexpr size_t SZ_WIN  = (size_t)3 * D_LAT * D_LAT * 2;                 // 6 MB
static constexpr size_t SZ_WOUT = (size_t)D_LAT * D_LAT * 2;                     // 2 MB
static constexpr size_t SZ_H    = (size_t)BATCH * 256 * D_LAT * 2;               // 64 MB
static constexpr size_t SZ_QKV  = (size_t)BATCH * 256 * 3 * D_LAT * 2;           // 192 MB
static constexpr size_t SZ_SC   = (size_t)BATCH * NHEAD * 256 * 256 * 4;         // 256 MB
static constexpr size_t SZ_ATT  = (size_t)BATCH * NHEAD * 256 * 256 * 2;         // 128 MB
static constexpr size_t SZ_O    = (size_t)BATCH * 256 * D_LAT * 2;               // 64 MB
static constexpr size_t SZ_PROJ = (size_t)BATCH * 256 * D_LAT * 4;               // 128 MB
static constexpr size_t SZ_ACC  = (size_t)BATCH * D_LAT * 4;                     // 512 KB

static constexpr size_t OFF_WIN  = 0;
static constexpr size_t OFF_WOUT = OFF_WIN  + SZ_WIN;
static constexpr size_t OFF_H    = OFF_WOUT + SZ_WOUT;
static constexpr size_t OFF_QKV  = OFF_H    + SZ_H;
static constexpr size_t OFF_SC   = OFF_QKV  + SZ_QKV;
static constexpr size_t OFF_ATT  = OFF_SC   + SZ_SC;
static constexpr size_t OFF_O    = OFF_ATT  + SZ_ATT;
static constexpr size_t OFF_PROJ = OFF_O    + SZ_O;
static constexpr size_t OFF_DP   = OFF_PROJ + SZ_PROJ;
static constexpr size_t OFF_TP   = OFF_DP   + SZ_ACC;

// ---------------------------------------------------------------------------
// One ParallelFcWithAttention module forward -> accum (B x D, f32)
// ---------------------------------------------------------------------------
static void run_module(const float* x, int xStride, int F,
                       const float* W, const float* bvec,
                       const float* in_w, const float* in_b,
                       const float* out_w, const float* out_b,
                       const float* g, const float* beta,
                       float* accum, uint8_t* ws, hipStream_t stream) {
  bf16_t* winb  = (bf16_t*)(ws + OFF_WIN);
  bf16_t* woutb = (bf16_t*)(ws + OFF_WOUT);
  bf16_t* hbuf  = (bf16_t*)(ws + OFF_H);
  bf16_t* qkv   = (bf16_t*)(ws + OFF_QKV);
  float*  sc    = (float*)(ws + OFF_SC);
  bf16_t* attn  = (bf16_t*)(ws + OFF_ATT);
  bf16_t* obuf  = (bf16_t*)(ws + OFF_O);
  float*  proj  = (float*)(ws + OFF_PROJ);

  const int Mrows = BATCH * F;
  const long qkvRowB = (long)F * 3 * D_LAT;   // per-batch stride into qkv
  const float invSqrtDh = 0.08838834764831845f;  // 1/sqrt(128)

  // weights -> bf16
  cvt_f32_bf16<<<256, 256, 0, stream>>>(in_w,  winb,  3 * D_LAT * D_LAT);
  cvt_f32_bf16<<<256, 256, 0, stream>>>(out_w, woutb, D_LAT * D_LAT);

  // h = relu(x * W + b), bf16
  feature_embed_relu<<<Mrows, 256, 0, stream>>>(x, xStride, F, W, bvec, hbuf);

  // qkv = h @ in_w^T + in_b   (M x 3D)
  {
    dim3 grid(3 * D_LAT / BN, Mrows / BM, 1);
    gemm_wmma<bf16_t, true><<<grid, 256, 0, stream>>>(
        hbuf, D_LAT, 0, 0,
        winb, D_LAT, 0, 0,
        qkv, 3 * D_LAT, 0, 0,
        in_b, 1.0f, Mrows, 3 * D_LAT, D_LAT, 1);
  }

  // scores = (q @ k^T) / sqrt(Dh)   per (b, head)
  {
    dim3 grid(F / BN, F / BM, BATCH * NHEAD);
    gemm_wmma<float, true><<<grid, 256, 0, stream>>>(
        qkv,          3 * D_LAT, qkvRowB, DHEAD,       // q
        qkv + D_LAT,  3 * D_LAT, qkvRowB, DHEAD,       // k (as N x K rows)
        sc, F, (long)NHEAD * F * F, (long)F * F,
        nullptr, invSqrtDh, F, F, DHEAD, NHEAD);
  }

  // softmax rows -> bf16 attn
  softmax_rows<<<BATCH * NHEAD * F, 256, 0, stream>>>(sc, attn, F);

  // o = attn @ v   (heads concatenated along D in obuf)
  {
    dim3 grid(DHEAD / BN, F / BM, BATCH * NHEAD);   // DHEAD == BN
    gemm_wmma<bf16_t, false><<<grid, 256, 0, stream>>>(
        attn, F, (long)NHEAD * F * F, (long)F * F,
        qkv + 2 * D_LAT, 3 * D_LAT, qkvRowB, DHEAD,  // v (K x N rows)
        obuf, D_LAT, (long)F * D_LAT, DHEAD,
        nullptr, 1.0f, F, DHEAD, F, NHEAD);
  }

  // proj = o @ out_w^T + out_b   (f32 for LayerNorm)
  {
    dim3 grid(D_LAT / BN, Mrows / BM, 1);
    gemm_wmma<float, true><<<grid, 256, 0, stream>>>(
        obuf, D_LAT, 0, 0,
        woutb, D_LAT, 0, 0,
        proj, D_LAT, 0, 0,
        out_b, 1.0f, Mrows, D_LAT, D_LAT, 1);
  }

  // LayerNorm + sum over F
  hipMemsetAsync(accum, 0, SZ_ACC, stream);
  layernorm_sum<<<Mrows, 256, 0, stream>>>(proj, g, beta, accum, F);
}

// ---------------------------------------------------------------------------
// Entry point
// ---------------------------------------------------------------------------
extern "C" void kernel_launch(void* const* d_in, const int* in_sizes, int n_in,
                              void* d_out, int out_size, void* d_ws, size_t ws_size,
                              hipStream_t stream) {
  (void)in_sizes; (void)n_in; (void)out_size; (void)ws_size;

  const float* drug    = (const float*)d_in[0];
  const float* target  = (const float*)d_in[1];
  const float* dW      = (const float*)d_in[2];
  const float* db      = (const float*)d_in[3];
  const float* d_in_w  = (const float*)d_in[4];
  const float* d_in_b  = (const float*)d_in[5];
  const float* d_out_w = (const float*)d_in[6];
  const float* d_out_b = (const float*)d_in[7];
  const float* d_g     = (const float*)d_in[8];
  const float* d_beta  = (const float*)d_in[9];
  const float* tW      = (const float*)d_in[10];
  const float* tb      = (const float*)d_in[11];
  const float* t_in_w  = (const float*)d_in[12];
  const float* t_in_b  = (const float*)d_in[13];
  const float* t_out_w = (const float*)d_in[14];
  const float* t_out_b = (const float*)d_in[15];
  const float* t_g     = (const float*)d_in[16];
  const float* t_beta  = (const float*)d_in[17];

  uint8_t* ws = (uint8_t*)d_ws;
  float* dp = (float*)(ws + OFF_DP);
  float* tp = (float*)(ws + OFF_TP);

  // drug module: F = 256, x row stride 2048
  run_module(drug, 2048, 256, dW, db, d_in_w, d_in_b, d_out_w, d_out_b,
             d_g, d_beta, dp, ws, stream);
  // target module: F = 128, x row stride 1024 (reuses all scratch)
  run_module(target, 1024, 128, tW, tb, t_in_w, t_in_b, t_out_w, t_out_b,
             t_g, t_beta, tp, ws, stream);

  cosine_sigmoid<<<BATCH, 256, 0, stream>>>(dp, tp, (float*)d_out);
}
